// QuadtreeLocalAttentionAggregator_26044681682972
// MI455X (gfx1250) — matrix operand order repaired and checked by reading, not compiled, not run on hardware
//
#include <hip/hip_runtime.h>
#include <hip/hip_bf16.h>

typedef __attribute__((ext_vector_type(16))) _Float16 v16h;
typedef __attribute__((ext_vector_type(8)))  float    v8f;
typedef __attribute__((ext_vector_type(4)))  _Float16 h4;
typedef __attribute__((ext_vector_type(2)))  _Float16 h2;

#define WMMA16(A,B,C) __builtin_amdgcn_wmma_f32_16x16x32_f16(false,(A),false,(B),(short)0,(C),false,false)

#define WPB 4                 // waves per block
#define RPW 16                // rows (supernodes) per wave, batched
#define BM_TOTAL (8 * 16384)  // 131072 rows
#define NMTX 8                // packed weight matrices

// ---------------------------------------------------------------------------
// Prologue: pack all eight 64x64 weight matrices (transposed use: B[k][n] =
// W[n][k]) into f16 WMMA B-fragment layout in d_ws.
//   B layout (CDNA5 16-bit B, 32x16 per fragment): lanes 0-15 hold K=0..7,16..23 as
//   pairs across 8 dwords, lanes 16-31 hold K=8..15,24..31; N = nt*16 + (lane&15).
//   Fragment order: wsH[(((mtx*2+kt)*4+nt)*32 + lane)*16 + 2j]
//   mtx: 0 Wsub | 1 wk | 2 wv | 3 Wq | 4 wq | 5 attn_ow | 6 Wout | 7 Wfb
// ---------------------------------------------------------------------------
__global__ void pack_b_frags(const float* __restrict__ Wsub,
                             const float* __restrict__ Wq,
                             const float* __restrict__ in_w,
                             const float* __restrict__ aow,
                             const float* __restrict__ Wout,
                             const float* __restrict__ Wfb,
                             _Float16* __restrict__ wsH) {
  int tid = blockIdx.x * blockDim.x + threadIdx.x;
  if (tid >= NMTX * 2 * 4 * 32 * 8) return;
  int j    = tid & 7;
  int lane = (tid >> 3) & 31;
  int nt   = (tid >> 8) & 3;
  int kt   = (tid >> 10) & 1;
  int mtx  = tid >> 11;                     // 0..7
  int n  = nt * 16 + (lane & 15);
  int k0 = kt * 32 + ((lane >> 4) << 3) + 2 * j + ((j >> 2) << 3);
  const float* src;
  switch (mtx) {
    case 0:  src = Wsub + n * 64;            break;  // kv_in = sub @ Wsub^T
    case 1:  src = in_w + (64 + n) * 64;     break;  // k = kv_in @ wk^T
    case 2:  src = in_w + (128 + n) * 64;    break;  // v = kv_in @ wv^T
    case 3:  src = Wq + n * 64;              break;  // q_in = super @ Wq^T
    case 4:  src = in_w + n * 64;            break;  // q = q_in @ wq^T
    case 5:  src = aow + n * 64;             break;  // attn_out = ctx @ aow^T
    case 6:  src = Wout + n * 64;            break;  // aggregated
    default: src = Wfb + n * 64;             break;  // fallback
  }
  int base = (((mtx * 2 + kt) * 4 + nt) * 32 + lane) * 16 + 2 * j;
  wsH[base]     = (_Float16)src[k0];
  wsH[base + 1] = (_Float16)src[k0 + 1];
}

// Read a 16x64 row-major f16 LDS tile as two K-tile A fragments (16-bit A
// layout: lane = M row (L&15), K pairs per dword, dwords 4-7 = K+16 block).
__device__ __forceinline__ void load_a_lds(const _Float16* buf, int m, int kh,
                                           v16h& a0, v16h& a1) {
  const _Float16* rp = buf + m * 64;
  #pragma unroll
  for (int j = 0; j < 4; ++j) {
    h2 p0 = *(const h2*)(rp + kh + 2 * j);       a0[2*j]   = p0[0]; a0[2*j+1] = p0[1];
    h2 p1 = *(const h2*)(rp + 16 + kh + 2 * j);  a0[8+2*j] = p1[0]; a0[9+2*j] = p1[1];
    h2 p2 = *(const h2*)(rp + 32 + kh + 2 * j);  a1[2*j]   = p2[0]; a1[2*j+1] = p2[1];
    h2 p3 = *(const h2*)(rp + 48 + kh + 2 * j);  a1[8+2*j] = p3[0]; a1[9+2*j] = p3[1];
  }
}

// D(16x64) = A(16x64) @ W[mtx]^T + bias, result stored row-major f16 in LDS.
// B fragments come straight from global d_ws (WGP$-resident, 32B/lane).
__device__ __forceinline__ void gemm16(const v16h& a0, const v16h& a1, int mtx,
                                       const float* __restrict__ bias,
                                       const _Float16* __restrict__ wsH,
                                       _Float16* dst, int L) {
  const int m = L & 15, mb = (L >> 4) << 3;
  #pragma unroll
  for (int nt = 0; nt < 4; ++nt) {
    v8f c = {};
    c = WMMA16(a0, *(const v16h*)(wsH + (mtx * 8 + nt) * 512 + L * 16), c);
    c = WMMA16(a1, *(const v16h*)(wsH + (mtx * 8 + 4 + nt) * 512 + L * 16), c);
    const int n = nt * 16 + m;
    const float b = bias[n];
    #pragma unroll
    for (int v = 0; v < 8; ++v)
      dst[(mb + v) * 64 + n] = (_Float16)(c[v] + b);
  }
}

__global__ __launch_bounds__(WPB * 32, 4)
void qlaa_kernel(const float* __restrict__ sub,  const int* __restrict__ mask,
                 const float* __restrict__ super,
                 const float* __restrict__ bsub, const float* __restrict__ bq,
                 const float* __restrict__ in_b, const float* __restrict__ aob,
                 const float* __restrict__ bout, const float* __restrict__ bfb,
                 const _Float16* __restrict__ wsH,
                 float* __restrict__ out) {
  __shared__ __align__(8) _Float16 XhA [WPB][1024];  // supernode 16x64 f16
  __shared__ __align__(8) _Float16 QhA [WPB][1024];  // q 16x64
  __shared__ __align__(8) _Float16 TmpA[WPB][1024];  // qin / kv_in / attn_out
  __shared__ __align__(8) _Float16 KhA [WPB][1024];  // k of current row
  __shared__ __align__(8) _Float16 VhA [WPB][1024];  // v of current row
  __shared__ __align__(8) _Float16 CtxA[WPB][1024];  // ctx rows 16x64
  __shared__ float scA[WPB][64];
  __shared__ float awA[WPB][64];
  __shared__ int   anyA[WPB][16];

  const int tid = threadIdx.x;
  const int w = tid >> 5, L = tid & 31;
  _Float16* Xh  = &XhA [w][0];
  _Float16* Qh  = &QhA [w][0];
  _Float16* Tmp = &TmpA[w][0];
  _Float16* Kh  = &KhA [w][0];
  _Float16* Vh  = &VhA [w][0];
  _Float16* Ctx = &CtxA[w][0];
  float* scp = &scA[w][0];
  float* awp = &awA[w][0];

  const int m  = L & 15;
  const int kh = (L >> 4) << 3;   // A-layout K half-offset (0 or 8)
  const int mb = kh;              // C/D-layout M base (0 or 8)
  const long r0 = ((long)blockIdx.x * WPB + w) * RPW;

  // ---- stage 16 supernode vectors (f32 -> f16), per-row mask flags ----
  {
    const float* supg = super + (size_t)r0 * 64;
    #pragma unroll
    for (int i = 0; i < 8; ++i) {
      int f4 = L + 32 * i;                       // 256 float4 = 16x64
      float4 p = ((const float4*)supg)[f4];
      h4 hh = { (_Float16)p.x, (_Float16)p.y, (_Float16)p.z, (_Float16)p.w };
      *(h4*)(Xh + f4 * 4) = hh;
    }
  }
  if (L < 16) {
    const int* mp = mask + (size_t)(r0 + L) * 16;
    int a = 0;
    #pragma unroll
    for (int jj = 0; jj < 16; ++jj) a |= mp[jj];
    anyA[w][L] = a;
  }
  __syncthreads();

  // ---- batched q path: Qin = X @ Wq^T + bq ; Q = Qin @ wq^T + bqi ----
  {
    v16h a0, a1;
    load_a_lds(Xh, m, kh, a0, a1);
    gemm16(a0, a1, 3, bq, wsH, Tmp, L);
    __syncthreads();
    load_a_lds(Tmp, m, kh, a0, a1);
    gemm16(a0, a1, 4, in_b, wsH, Qh, L);
    __syncthreads();
  }

  // ---- per-row: kv/k/v GEMMs + masked softmax attention -> ctx row ----
  for (int rr = 0; rr < RPW; ++rr) {
    const long r = r0 + rr;
    const float* arow = sub + ((size_t)r * 16 + m) * 64;
    const int*   mrow = mask + (size_t)r * 16;

    v16h s0, s1;                                  // A frags of subnode tile
    #pragma unroll
    for (int j = 0; j < 4; ++j) {
      float2 p0 = *(const float2*)(arow + kh + 2 * j);
      s0[2*j]   = (_Float16)p0.x; s0[2*j+1] = (_Float16)p0.y;
      float2 p1 = *(const float2*)(arow + 16 + kh + 2 * j);
      s0[8+2*j] = (_Float16)p1.x; s0[9+2*j] = (_Float16)p1.y;
      float2 p2 = *(const float2*)(arow + 32 + kh + 2 * j);
      s1[2*j]   = (_Float16)p2.x; s1[2*j+1] = (_Float16)p2.y;
      float2 p3 = *(const float2*)(arow + 48 + kh + 2 * j);
      s1[8+2*j] = (_Float16)p3.x; s1[9+2*j] = (_Float16)p3.y;
    }
    gemm16(s0, s1, 0, bsub, wsH, Tmp, L);         // kv_in
    __syncthreads();
    v16h kv0, kv1;
    load_a_lds(Tmp, m, kh, kv0, kv1);
    gemm16(kv0, kv1, 1, in_b + 64,  wsH, Kh, L);  // k
    gemm16(kv0, kv1, 2, in_b + 128, wsH, Vh, L);  // v
    __syncthreads();

    // scores[h][n] = 0.25 * q[h,:] . k[n,h,:], masked
    #pragma unroll
    for (int half = 0; half < 2; ++half) {
      const int idx = L + 32 * half;
      const int h = idx >> 4, mm = idx & 15;
      float s = 0.f;
      #pragma unroll
      for (int d = 0; d < 16; ++d)
        s += (float)Qh[rr * 64 + h * 16 + d] * (float)Kh[mm * 64 + h * 16 + d];
      s *= 0.25f;
      scp[idx] = (mrow[mm] != 0) ? s : -1.0e9f;
    }
    __syncthreads();

    // softmax over 16 subnodes per head
    #pragma unroll
    for (int half = 0; half < 2; ++half) {
      const int idx = L + 32 * half;
      const int h = idx >> 4;
      float mx = -3.4e38f;
      #pragma unroll
      for (int m2 = 0; m2 < 16; ++m2) mx = fmaxf(mx, scp[h * 16 + m2]);
      float den = 0.f;
      #pragma unroll
      for (int m2 = 0; m2 < 16; ++m2) den += __expf(scp[h * 16 + m2] - mx);
      awp[idx] = __expf(scp[idx] - mx) / den;
    }
    __syncthreads();

    // ctx[h][d] = sum_n attn[h][n] * v[n,h,d] ; E index = h*16+d = idx
    #pragma unroll
    for (int half = 0; half < 2; ++half) {
      const int idx = L + 32 * half;
      const int h = idx >> 4, d = idx & 15;
      float acc = 0.f;
      #pragma unroll
      for (int m2 = 0; m2 < 16; ++m2)
        acc += awp[h * 16 + m2] * (float)Vh[m2 * 64 + h * 16 + d];
      Ctx[rr * 64 + idx] = (_Float16)acc;
    }
    __syncthreads();
  }

  // ---- batched out path: AO = Ctx@aow^T+aob; AGG = AO@Wout^T+bout;
  //      FB = X@Wfb^T+bfb; masked select fused with writeout ----
  {
    v16h c0, c1;
    load_a_lds(Ctx, m, kh, c0, c1);
    gemm16(c0, c1, 5, aob, wsH, Tmp, L);          // attn_out
    __syncthreads();
    v16h o0, o1, x0, x1;
    load_a_lds(Tmp, m, kh, o0, o1);
    load_a_lds(Xh, m, kh, x0, x1);
    #pragma unroll
    for (int nt = 0; nt < 4; ++nt) {
      v8f cg = {}, cf = {};
      cg = WMMA16(o0, *(const v16h*)(wsH + (6 * 8 + nt) * 512 + L * 16), cg);
      cg = WMMA16(o1, *(const v16h*)(wsH + (6 * 8 + 4 + nt) * 512 + L * 16), cg);
      cf = WMMA16(x0, *(const v16h*)(wsH + (7 * 8 + nt) * 512 + L * 16), cf);
      cf = WMMA16(x1, *(const v16h*)(wsH + (7 * 8 + 4 + nt) * 512 + L * 16), cf);
      const int n = nt * 16 + m;
      const float bg = bout[n], bf2 = bfb[n];
      #pragma unroll
      for (int v = 0; v < 8; ++v) {
        const int row = mb + v;
        const float val = (anyA[w][row] != 0) ? (cg[v] + bg) : (cf[v] + bf2);
        out[(size_t)(r0 + row) * 64 + n] = val;
      }
    }
  }
}

extern "C" void kernel_launch(void* const* d_in, const int* in_sizes, int n_in,
                              void* d_out, int out_size, void* d_ws, size_t ws_size,
                              hipStream_t stream) {
  const float* sub   = (const float*)d_in[0];
  const int*   mask  = (const int*)d_in[1];
  const float* super = (const float*)d_in[2];
  const float* Wsub  = (const float*)d_in[3];
  const float* bsub  = (const float*)d_in[4];
  const float* Wq    = (const float*)d_in[5];
  const float* bq    = (const float*)d_in[6];
  const float* in_w  = (const float*)d_in[7];
  const float* in_b  = (const float*)d_in[8];
  const float* aow   = (const float*)d_in[9];
  const float* aob   = (const float*)d_in[10];
  const float* Wout  = (const float*)d_in[11];
  const float* bout  = (const float*)d_in[12];
  const float* Wfb   = (const float*)d_in[13];
  const float* bfb   = (const float*)d_in[14];
  _Float16* wsH = (_Float16*)d_ws;     // 8*2*4*32*16 halves = 64 KB
  float* outp = (float*)d_out;

  pack_b_frags<<<64, 256, 0, stream>>>(Wsub, Wq, in_w, aow, Wout, Wfb, wsH);

  const int blocks = BM_TOTAL / (WPB * RPW);  // 2048
  qlaa_kernel<<<blocks, WPB * 32, 0, stream>>>(
      sub, mask, super, bsub, bq, in_b, aob, bout, bfb, wsH, outp);
}
